// LSTMBaseline_20598663152193
// MI455X (gfx1250) — matrix operand order repaired and checked by reading, not compiled, not run on hardware
//
#include <hip/hip_runtime.h>
#include <math.h>

// ---------------------------------------------------------------------------
// MI455X (gfx1250, wave32) fused 2-layer LSTM step + decoder.
//  * Gate GEMMs: v_wmma_f32_16x16x32_bf16, f-gate and W_hh skipped (zero state)
//  * A tiles staged in LDS via global_load_async_to_lds_b128 (double-buffered,
//    s_wait_asynccnt), shared by the 4 waves of a block.
//  * B fragments software-pipelined in registers across K iterations.
//  * Activations via v_exp_f32 / v_rcp_f32 (no libm tanhf).
// ---------------------------------------------------------------------------

typedef __attribute__((ext_vector_type(8)))  __bf16 v8bf;
typedef __attribute__((ext_vector_type(16))) __bf16 v16bf;
typedef __attribute__((ext_vector_type(8)))  float  v8f;

#define LSTM_B 16384
#define LSTM_D 512
#define LSTM_H 1024

#define TILE_ROWS 64      // A-tile rows per block
#define ROW_PITCH 80      // padded LDS row pitch (bank-conflict-free b128 reads)

static __device__ __forceinline__ unsigned short f32_to_bf16_rne(float f) {
  unsigned int u = __float_as_uint(f);
  u += 0x7FFFu + ((u >> 16) & 1u);
  return (unsigned short)(u >> 16);
}

// Fast activations: v_exp_f32 + v_rcp_f32; saturate correctly via inf.
static __device__ __forceinline__ float sigmoid_fast(float x) {
  return __builtin_amdgcn_rcpf(1.0f + __expf(-x));
}
static __device__ __forceinline__ float tanh_fast(float x) {
  return 1.0f - 2.0f * __builtin_amdgcn_rcpf(1.0f + __expf(2.0f * x));
}

// Raw LDS byte offset of a __shared__ object (generic -> AS(3) -> int).
static __device__ __forceinline__ unsigned lds_addr_of(const void* p) {
  return (unsigned)(size_t)(__attribute__((address_space(3))) const void*)p;
}

// f32 -> bf16 conversion, 4 elems/thread. n must be a multiple of 4.
__global__ __launch_bounds__(256)
void cvt_f32_bf16(const float* __restrict__ src,
                  unsigned short* __restrict__ dst, int n) {
  int i = (blockIdx.x * blockDim.x + threadIdx.x) * 4;
  if (i + 3 < n) {
    float4 v = *(const float4*)(src + i);
    ushort2 lo = { f32_to_bf16_rne(v.x), f32_to_bf16_rne(v.y) };
    ushort2 hi = { f32_to_bf16_rne(v.z), f32_to_bf16_rne(v.w) };
    *(ushort2*)(dst + i)     = lo;
    *(ushort2*)(dst + i + 2) = hi;
  }
}

// Issue async copy of a 64x32 bf16 A tile (global, row pitch gpitch bytes)
// into LDS at byte offset lbase (row pitch ROW_PITCH). 128 threads, 2 b128
// lane-transfers each => 2 async instructions per wave (ASYNCcnt += 2).
static __device__ __forceinline__ void async_copy_tileA(
    const unsigned char* __restrict__ gbase, size_t gpitch,
    unsigned lbase, int tid) {
  #pragma unroll
  for (int p = 0; p < 2; ++p) {
    const int idx   = tid + p * 128;          // 0..255
    const int row   = idx >> 2;               // 0..63
    const int chunk = idx & 3;                // 16B chunk within row
    const unsigned long long ga =
        (unsigned long long)(gbase + (size_t)row * gpitch + chunk * 16);
    const unsigned la = lbase + (unsigned)(row * ROW_PITCH + chunk * 16);
    asm volatile("global_load_async_to_lds_b128 %0, %1, off"
                 :: "v"(la), "v"(ga) : "memory");
  }
}

// A fragment (16x32 bf16 MxK) from the LDS tile: lane holds row (lane&15) of
// the 16-row subtile; K chunks at bytes 16*half and 32+16*half.
static __device__ __forceinline__ v16bf lds_frag_a(const unsigned char* tile,
                                                   int mt, int lane) {
  const int half = lane >> 4;
  const unsigned char* p = tile + (mt * 16 + (lane & 15)) * ROW_PITCH + 16 * half;
  v8bf lo = *(const v8bf*)p;
  v8bf hi = *(const v8bf*)(p + 32);
  return __builtin_shufflevector(lo, hi, 0,1,2,3,4,5,6,7,8,9,10,11,12,13,14,15);
}

// B fragment (32x16 bf16 KxN) from row-major W (N x K): lane holds W-row
// (lane&15); K chunks at 8*(lane>>4) and 16+8*(lane>>4).
static __device__ __forceinline__ v16bf load_frag_b(const __bf16* __restrict__ w,
                                                    int ld, int n0, int k0, int lane) {
  const int half = lane >> 4;
  const __bf16* p = w + (size_t)(n0 + (lane & 15)) * ld + k0 + 8 * half;
  v8bf lo = *(const v8bf*)p;
  v8bf hi = *(const v8bf*)(p + 16);
  return __builtin_shufflevector(lo, hi, 0,1,2,3,4,5,6,7,8,9,10,11,12,13,14,15);
}

// Fused LSTM layer (single step, zero state):
//   h[m,j] = sig(o)*tanh( sig(i)*tanh(g) ),  gate = A @ W_ih[gate-rows]^T + b
// Block = 4 waves; shared 64-row A tile (LDS, double-buffered async); each
// wave owns 16 h-columns with 3 accumulator groups (i,g,o).
__global__ __launch_bounds__(128)
void lstm_layer_wmma(const __bf16* __restrict__ A,     // (M, K) bf16 row-major
                     const __bf16* __restrict__ W,     // (4H, K) bf16 row-major
                     const float*  __restrict__ b_ih,  // (4H,)
                     const float*  __restrict__ b_hh,  // (4H,)
                     unsigned short* __restrict__ Hout,// (M, H) bf16
                     int K, int M) {
  const int H    = LSTM_H;
  const int tid  = threadIdx.x;
  const int wave = tid >> 5;
  const int lane = tid & 31;
  const int m0   = blockIdx.x * TILE_ROWS;
  const int j0   = blockIdx.y * 64 + wave * 16;

  __shared__ __align__(16) unsigned char tileA[2][TILE_ROWS * ROW_PITCH];
  const unsigned lb[2] = { lds_addr_of(&tileA[0][0]), lds_addr_of(&tileA[1][0]) };

  const int goff[3] = {0, 2 * H, 3 * H};   // i, g, o gate rows (f skipped)
  const size_t gpitch = (size_t)K * 2;     // A row pitch in bytes
  const unsigned char* Abytes =
      (const unsigned char*)A + (size_t)m0 * gpitch;

  v8f acc[4][3];
  #pragma unroll
  for (int mt = 0; mt < 4; ++mt)
    #pragma unroll
    for (int g = 0; g < 3; ++g)
      acc[mt][g] = (v8f){0.f,0.f,0.f,0.f,0.f,0.f,0.f,0.f};

  const int nk = K / 32;

  // Prologue: async A tile 0 + B fragments for kt=0 (register pipeline).
  async_copy_tileA(Abytes, gpitch, lb[0], tid);
  v16bf bfrag[3];
  #pragma unroll
  for (int g = 0; g < 3; ++g)
    bfrag[g] = load_frag_b(W, K, goff[g] + j0, 0, lane);

  for (int kt = 0; kt < nk; ++kt) {
    const int cur = kt & 1;
    if (kt + 1 < nk) {
      // Prefetch next A tile; buffer cur^1 was drained by last iter's barrier.
      async_copy_tileA(Abytes + (size_t)(kt + 1) * 64, gpitch, lb[cur ^ 1], tid);
      asm volatile("s_wait_asynccnt 0x2" ::: "memory");  // current tile landed
    } else {
      asm volatile("s_wait_asynccnt 0x0" ::: "memory");
    }
    __syncthreads();                                     // whole tile visible

    // Issue next-iteration B loads first (long latency, consumed next iter).
    v16bf bnext[3];
    if (kt + 1 < nk) {
      #pragma unroll
      for (int g = 0; g < 3; ++g)
        bnext[g] = load_frag_b(W, K, goff[g] + j0, (kt + 1) * 32, lane);
    }

    // A fragments as one ds clause (short latency, consumed now).
    v16bf af[4];
    #pragma unroll
    for (int mt = 0; mt < 4; ++mt)
      af[mt] = lds_frag_a(&tileA[cur][0], mt, lane);

    #pragma unroll
    for (int mt = 0; mt < 4; ++mt)
      #pragma unroll
      for (int g = 0; g < 3; ++g)
        acc[mt][g] = __builtin_amdgcn_wmma_f32_16x16x32_bf16(
            false, af[mt], false, bfrag[g], (short)0, acc[mt][g], false, false);

    __syncthreads();                                     // done reading 'cur'
    if (kt + 1 < nk) {
      #pragma unroll
      for (int g = 0; g < 3; ++g)
        bfrag[g] = bnext[g];
    }
  }

  // Epilogue: bias + fast activations, write bf16 h.
  const int col  = j0 + (lane & 15);
  const int half = lane >> 4;
  const float bi = b_ih[col]         + b_hh[col];
  const float bg = b_ih[2 * H + col] + b_hh[2 * H + col];
  const float bo = b_ih[3 * H + col] + b_hh[3 * H + col];

  #pragma unroll
  for (int mt = 0; mt < 4; ++mt) {
    #pragma unroll
    for (int j = 0; j < 8; ++j) {
      const int row = m0 + mt * 16 + j + 8 * half;
      const float iv = acc[mt][0][j] + bi;
      const float gv = acc[mt][1][j] + bg;
      const float ov = acc[mt][2][j] + bo;
      const float c  = sigmoid_fast(iv) * tanh_fast(gv);
      const float h  = sigmoid_fast(ov) * tanh_fast(c);
      Hout[(size_t)row * H + col] = f32_to_bf16_rne(h);
    }
  }
}

// Decoder: out = h1 @ W_dec^T + b_dec, fp32 output. Wave tile 64x16.
__global__ __launch_bounds__(128)
void decoder_wmma(const __bf16* __restrict__ A,   // (M, K) bf16
                  const __bf16* __restrict__ W,   // (D, K) bf16
                  const float*  __restrict__ bias,// (D,)
                  float* __restrict__ Out,        // (M, D) fp32
                  int K, int M) {
  const int D    = LSTM_D;
  const int wave = threadIdx.x >> 5;
  const int lane = threadIdx.x & 31;
  const int m0   = blockIdx.x * 64;
  const int n0   = blockIdx.y * 64 + wave * 16;
  const int half = lane >> 4;

  v8f acc[4];
  #pragma unroll
  for (int mt = 0; mt < 4; ++mt)
    acc[mt] = (v8f){0.f,0.f,0.f,0.f,0.f,0.f,0.f,0.f};

  for (int k0 = 0; k0 < K; k0 += 32) {
    v16bf af[4];
    #pragma unroll
    for (int mt = 0; mt < 4; ++mt) {
      const __bf16* p = A + (size_t)(m0 + mt * 16 + (lane & 15)) * K + k0 + 8 * half;
      v8bf lo = *(const v8bf*)p;
      v8bf hi = *(const v8bf*)(p + 16);
      af[mt] = __builtin_shufflevector(lo, hi, 0,1,2,3,4,5,6,7,8,9,10,11,12,13,14,15);
    }
    v16bf bfrag = load_frag_b(W, K, n0, k0, lane);
    #pragma unroll
    for (int mt = 0; mt < 4; ++mt)
      acc[mt] = __builtin_amdgcn_wmma_f32_16x16x32_bf16(
          false, af[mt], false, bfrag, (short)0, acc[mt], false, false);
  }

  const int col = n0 + (lane & 15);
  const float b = bias[col];
  #pragma unroll
  for (int mt = 0; mt < 4; ++mt) {
    #pragma unroll
    for (int j = 0; j < 8; ++j) {
      const int row = m0 + mt * 16 + j + 8 * half;
      Out[(size_t)row * D + col] = acc[mt][j] + b;
    }
  }
}

extern "C" void kernel_launch(void* const* d_in, const int* in_sizes, int n_in,
                              void* d_out, int out_size, void* d_ws, size_t ws_size,
                              hipStream_t stream) {
  (void)in_sizes; (void)n_in; (void)out_size; (void)ws_size;
  const float* x     = (const float*)d_in[0];
  const float* W_ih0 = (const float*)d_in[1];
  // d_in[2] = W_hh0 : unused (h0 = 0)
  const float* b_ih0 = (const float*)d_in[3];
  const float* b_hh0 = (const float*)d_in[4];
  const float* W_ih1 = (const float*)d_in[5];
  // d_in[6] = W_hh1 : unused
  const float* b_ih1 = (const float*)d_in[7];
  const float* b_hh1 = (const float*)d_in[8];
  const float* W_dec = (const float*)d_in[9];
  const float* b_dec = (const float*)d_in[10];
  float* out = (float*)d_out;

  // Workspace layout (all 256B aligned)
  char* ws = (char*)d_ws;
  unsigned short* xb  = (unsigned short*)(ws);                         // 16384x512  = 16 MiB
  unsigned short* w0b = (unsigned short*)(ws + 16777216);              // 4096x512   =  4 MiB
  unsigned short* w1b = (unsigned short*)(ws + 16777216 + 4194304);    // 4096x1024  =  8 MiB
  unsigned short* wdb = (unsigned short*)(ws + 16777216 + 4194304 + 8388608);            // 512x1024
  unsigned short* h0b = (unsigned short*)(ws + 16777216 + 4194304 + 8388608 + 1048576);  // 16384x1024
  unsigned short* h1b = (unsigned short*)(ws + 16777216 + 4194304 + 8388608 + 1048576 + 33554432);

  // bf16 conversions
  {
    const int n_x  = LSTM_B * LSTM_D;       // 8388608
    const int n_w0 = 4 * LSTM_H * LSTM_D;   // 2097152
    const int n_w1 = 4 * LSTM_H * LSTM_H;   // 4194304
    const int n_wd = LSTM_D * LSTM_H;       // 524288
    cvt_f32_bf16<<<dim3(n_x  / 4 / 256), dim3(256), 0, stream>>>(x,     xb,  n_x);
    cvt_f32_bf16<<<dim3(n_w0 / 4 / 256), dim3(256), 0, stream>>>(W_ih0, w0b, n_w0);
    cvt_f32_bf16<<<dim3(n_w1 / 4 / 256), dim3(256), 0, stream>>>(W_ih1, w1b, n_w1);
    cvt_f32_bf16<<<dim3(n_wd / 4 / 256), dim3(256), 0, stream>>>(W_dec, wdb, n_wd);
  }

  const dim3 blk(128);
  // Layer 0: A = x bf16 (16384x512), W_ih0 (4096x512)
  lstm_layer_wmma<<<dim3(LSTM_B / 64, LSTM_H / 64), blk, 0, stream>>>(
      (const __bf16*)xb, (const __bf16*)w0b, b_ih0, b_hh0, h0b, LSTM_D, LSTM_B);
  // Layer 1: A = h0 bf16 (16384x1024), W_ih1 (4096x1024)
  lstm_layer_wmma<<<dim3(LSTM_B / 64, LSTM_H / 64), blk, 0, stream>>>(
      (const __bf16*)h0b, (const __bf16*)w1b, b_ih1, b_hh1, h1b, LSTM_H, LSTM_B);
  // Decoder: out = h1 @ W_dec^T + b_dec (fp32)
  decoder_wmma<<<dim3(LSTM_B / 64, LSTM_D / 64), blk, 0, stream>>>(
      (const __bf16*)h1b, (const __bf16*)wdb, b_dec, out, LSTM_H, LSTM_B);
}
